// TransVLAD_57758720197097
// MI455X (gfx1250) — matrix operand order repaired and checked by reading, not compiled, not run on hardware
//
#include <hip/hip_runtime.h>
#include <hip/hip_bf16.h>
#include <stdint.h>

// ---------------------------------------------------------------------------
// TransVLAD for MI455X (gfx1250, wave32, WMMA, async LDS staging).
// bf16 operands, f32 accumulation via v_wmma_f32_16x16x32_bf16.
// ---------------------------------------------------------------------------

typedef __bf16 bf16_t;
typedef __attribute__((ext_vector_type(16))) __bf16 v16bf;
typedef __attribute__((ext_vector_type(8)))  __bf16 v8bf;
typedef __attribute__((ext_vector_type(8)))  float  v8f;
typedef __attribute__((ext_vector_type(4)))  int    v4i;

#define NB 32      // batch
#define CC 512     // input channels
#define LL 1200    // H*W
#define LP 1216    // L padded to multiple of 64 (zero-filled tail)
#define EE 1024    // expansion dim
#define GG 8       // groups
#define KK 64      // clusters
#define DG 128     // per-group dim

// ---------------------------------------------------------------------------
// Async global -> LDS staging (CDNA5 GLOBAL_LOAD_ASYNC_TO_LDS_B128, ASYNCcnt)
// Probe-learned signature: (addrspace(1) v4i*, addrspace(3) v4i*, imm, imm).
// Guarded: falls back to a synchronous 16B copy if the builtin is absent.
// ---------------------------------------------------------------------------
#if __has_builtin(__builtin_amdgcn_global_load_async_to_lds_b128) && \
    __has_builtin(__builtin_amdgcn_s_wait_asynccnt)
#define ASYNC_LDS 1
#endif

__device__ __forceinline__ void g2l_b128(void* lds, const void* g) {
#ifdef ASYNC_LDS
  __builtin_amdgcn_global_load_async_to_lds_b128(
      (__attribute__((address_space(1))) v4i*)(g),
      (__attribute__((address_space(3))) v4i*)(lds), 0, 0);
#else
  *(uint4*)lds = *(const uint4*)g;
#endif
}
__device__ __forceinline__ void wait_async_lds() {
#ifdef ASYNC_LDS
  __builtin_amdgcn_s_wait_asynccnt(0);
#endif
}

// ---------------------------------------------------------------------------
// WMMA helpers (CDNA5 ISA §7.12.2, wave32 layouts).
// ---------------------------------------------------------------------------
__device__ __forceinline__ v8f wmma_bf16(v16bf a, v16bf b, v8f c) {
  return __builtin_amdgcn_wmma_f32_16x16x32_bf16(
      false, a, false, b, (short)0, c, false, false);
}

// A 16x32 bf16: lane holds row m = lane&15; elements 0-7 = K klo..klo+7,
// elements 8-15 = K klo+16..klo+23, klo = (lane<16 ? 0 : 8).
// LDS tile must be row-major over K with stride ld; loads = 2x ds_load_b128.
__device__ __forceinline__ v16bf load_frag_a(const bf16_t* rowbase, int ld) {
  const int lane = threadIdx.x & 31;
  const bf16_t* p = rowbase + (lane & 15) * ld + ((lane < 16) ? 0 : 8);
  v8bf c0 = *(const v8bf*)__builtin_assume_aligned(p, 16);
  v8bf c1 = *(const v8bf*)__builtin_assume_aligned(p + 16, 16);
  union { v16bf v; v8bf h[2]; } u;
  u.h[0] = c0; u.h[1] = c1;
  return u.v;
}

// B 32x16 bf16: lane holds column n = lane&15; elements 0-7 = K klo..klo+7,
// elements 8-15 = K klo+16..klo+23, klo = (lane<16 ? 0 : 8). LDS tile stored column-major over the
// contraction dim (tile[n][K], stride ld); loads = 2x ds_load_b128.
__device__ __forceinline__ v16bf load_frag_b(const bf16_t* colbase, int ld) {
  const int lane = threadIdx.x & 31;
  const bf16_t* p = colbase + (lane & 15) * ld + ((lane < 16) ? 0 : 8);
  v8bf c0 = *(const v8bf*)__builtin_assume_aligned(p, 16);
  v8bf c1 = *(const v8bf*)__builtin_assume_aligned(p + 16, 16);
  union { v16bf v; v8bf h[2]; } u;
  u.h[0] = c0; u.h[1] = c1;
  return u.v;
}

// D/C 16x16 f32: element r -> row = r + (lane<16?0:8), col = lane&15.

// ---------------------------------------------------------------------------
// Kernel 0: fp32 -> bf16 weight conversion
// ---------------------------------------------------------------------------
__global__ void k_cvt_bf16(const float* __restrict__ in,
                           bf16_t* __restrict__ out, int count) {
  int i = blockIdx.x * blockDim.x + threadIdx.x;
  if (i < count) out[i] = (bf16_t)in[i];
}

// ---------------------------------------------------------------------------
// Kernel 1: L2 normalize over C, write TRANSPOSED xhatT[n][l][c] (c contig),
// zero-fill padded rows l in [LL, LP).
// ---------------------------------------------------------------------------
__global__ __launch_bounds__(256)
void k_normalize(const float* __restrict__ x, bf16_t* __restrict__ xhatT) {
  const int n = blockIdx.y;
  const int l = blockIdx.x * 256 + threadIdx.x;
  if (l >= LP) return;
  bf16_t* op = xhatT + ((size_t)n * LP + l) * CC;
  if (l >= LL) {
    v8bf z;
#pragma unroll
    for (int i = 0; i < 8; ++i) z[i] = (bf16_t)0.f;
    for (int c = 0; c < CC; c += 8) *(v8bf*)(op + c) = z;
    return;
  }
  const float* xp = x + (size_t)n * CC * LL + l;
  float s = 0.f;
#pragma unroll 8
  for (int c = 0; c < CC; ++c) {
    float v = xp[(size_t)c * LL];
    s += v * v;
  }
  const float inv = 1.0f / fmaxf(sqrtf(s), 1e-12f);
  for (int c0 = 0; c0 < CC; c0 += 8) {
    v8bf o;
#pragma unroll
    for (int i = 0; i < 8; ++i)
      o[i] = (bf16_t)(xp[(size_t)(c0 + i) * LL] * inv);
    *(v8bf*)(op + c0) = o;
  }
}

// ---------------------------------------------------------------------------
// Kernel 2: GEMM1  y = W1(1024x512) @ xhat(512xL)  ->  ybT[n][l][e] bf16.
// 128(M=e) x 64(N=l) tile, K-step 32. 256 thr = 8 waves; wave = one m-tile,
// 4 n-tile f32 accumulators. Tiles staged with async b128 copies.
// ---------------------------------------------------------------------------
__global__ __launch_bounds__(256)
void k_gemm1(const bf16_t* __restrict__ W1b, const bf16_t* __restrict__ xhatT,
             bf16_t* __restrict__ ybT) {
  __shared__ __align__(16) bf16_t As[128][32];  // [e][c]  8 KB
  __shared__ __align__(16) bf16_t Bt[64][32];   // [l][c]  4 KB (c contig!)

  const int n = blockIdx.z;
  const int e0 = blockIdx.y * 128;
  const int l0 = blockIdx.x * 64;
  const int tid = threadIdx.x, lane = tid & 31, wid = tid >> 5;

  v8f acc[4] = {{}, {}, {}, {}};
  const bf16_t* xb = xhatT + (size_t)n * LP * CC;

  for (int k0 = 0; k0 < CC; k0 += 32) {
    // A tile: 128 rows x 64B = 512 x 16B chunks -> 2 per thread
#pragma unroll
    for (int i = tid; i < 512; i += 256) {
      int r = i >> 2, cc = (i & 3) * 8;
      g2l_b128(&As[r][cc], W1b + (size_t)(e0 + r) * CC + k0 + cc);
    }
    // B tile: 64 rows x 64B = 256 x 16B chunks -> 1 per thread
    {
      int r = tid >> 2, cc = (tid & 3) * 8;
      g2l_b128(&Bt[r][cc], xb + (size_t)(l0 + r) * CC + k0 + cc);
    }
    if (k0 + 32 < CC)  // global_prefetch_b8 of next B slice
      __builtin_prefetch(xb + (size_t)(l0 + (tid >> 2)) * CC + k0 + 32 + (tid & 3) * 8, 0, 1);
    wait_async_lds();
    __syncthreads();

    v16bf a = load_frag_a(&As[wid * 16][0], 32);
#pragma unroll
    for (int nt = 0; nt < 4; ++nt) {
      v16bf b = load_frag_b(&Bt[nt * 16][0], 32);
      acc[nt] = wmma_bf16(a, b, acc[nt]);
    }
    __syncthreads();
  }

  // store: per lane, 8 consecutive e -> one 16B bf16 vector store per n-tile
  const int eoff = e0 + wid * 16 + ((lane < 16) ? 0 : 8);
#pragma unroll
  for (int nt = 0; nt < 4; ++nt) {
    int l = l0 + nt * 16 + (lane & 15);
    v8bf o;
#pragma unroll
    for (int r = 0; r < 8; ++r) o[r] = (bf16_t)acc[nt][r];
    *(v8bf*)(ybT + ((size_t)n * LP + l) * EE + eoff) = o;
  }
}

// ---------------------------------------------------------------------------
// Kernel 3: attention att[n,g,l] = sigmoid(sum_e W2[g,e] * y[n,e,l]).
// Tiny: VALU with W2 in LDS; y rows are contiguous in ybT -> b128 loads.
// ---------------------------------------------------------------------------
__global__ __launch_bounds__(256)
void k_att(const bf16_t* __restrict__ ybT, const float* __restrict__ W2,
           float* __restrict__ att) {
  __shared__ float w2s[GG * EE];  // 32 KB
  const int tid = threadIdx.x;
  for (int i = tid; i < GG * EE; i += 256) w2s[i] = W2[i];
  __syncthreads();

  const int n = blockIdx.y;
  const int l = blockIdx.x * 256 + tid;
  if (l >= LL) return;

  float acc[GG];
#pragma unroll
  for (int g = 0; g < GG; ++g) acc[g] = 0.f;

  const bf16_t* yp = ybT + ((size_t)n * LP + l) * EE;
  for (int e0 = 0; e0 < EE; e0 += 8) {
    v8bf c = *(const v8bf*)__builtin_assume_aligned(yp + e0, 16);
#pragma unroll
    for (int i = 0; i < 8; ++i) {
      float v = (float)c[i];
#pragma unroll
      for (int g = 0; g < GG; ++g) acc[g] += w2s[g * EE + e0 + i] * v;
    }
  }
#pragma unroll
  for (int g = 0; g < GG; ++g)
    att[((size_t)n * GG + g) * LL + l] = 1.0f / (1.0f + __expf(-acc[g]));
}

// ---------------------------------------------------------------------------
// Kernel 4: fused logits -> softmax -> attention weighting -> VLAD reduce.
// One 1024-thread WG (32 wave32) per batch n; wave owns (dt,kt) 16x16 tile
// of the 128(d) x 64(k) accumulation.
// ---------------------------------------------------------------------------
#define VT 32
__global__ __launch_bounds__(1024)
void k_vlad(const bf16_t* __restrict__ ybT, const bf16_t* __restrict__ Wcb,
            const float* __restrict__ att, const float* __restrict__ centroids,
            float* __restrict__ out) {
  __shared__ __align__(16) bf16_t wc_s[KK][DG];    // 16 KB  [k][d] (A, stage1)
  __shared__ __align__(16) bf16_t xgT[VT][DG];     //  8 KB  [j][d] (B, stage1)
  __shared__ __align__(16) bf16_t xg2[DG][VT];     //  8 KB  [d][j] (A, stage2)
  __shared__ __align__(16) float  S_s[KK][VT + 1]; // ~8.4 KB logits
  __shared__ __align__(16) bf16_t w_s[KK][VT];     //  4 KB  [k][j] (B, stage2)
  __shared__ float cmax[VT], cscale[VT];
  __shared__ float wsum_s[KK];

  const int n = blockIdx.x;
  const int tid = threadIdx.x, lane = tid & 31, wid = tid >> 5;
  const int dt = wid >> 2, kt = wid & 3;
  const int myk = tid >> 5, myj = tid & 31;

  // async-stage cluster weights [k][d] (16 KB = 1024 x 16B, 1 per thread)
  {
    int r = tid >> 4, cc = (tid & 15) * 8;
    g2l_b128(&wc_s[r][cc], Wcb + (size_t)r * DG + cc);
  }
  if (tid < KK) wsum_s[tid] = 0.f;

  v8f acc = {};
  float ws0 = 0.f, ws1 = 0.f;
  const bf16_t* ybn = ybT + (size_t)n * LP * EE;

  for (int g = 0; g < GG; ++g) {
    const int gbase = g * DG;
    const float* ap = att + ((size_t)n * GG + g) * LL;

    for (int lt = 0; lt < LP / VT; ++lt) {
      const int l0 = lt * VT;

      // ---- stage xgT[j][d] tile: 32 rows x 256B = 512 chunks, tid<512 ----
      if (tid < 512) {
        int j = tid >> 4, cc = (tid & 15) * 8;
        g2l_b128(&xgT[j][cc], ybn + (size_t)(l0 + j) * EE + gbase + cc);
      }
      wait_async_lds();
      __syncthreads();

      // ---- build xg2[d][j] transpose copy (4 elems per thread) ----
      {
        int j = tid >> 5, d0 = (tid & 31) * 4;
#pragma unroll
        for (int i = 0; i < 4; ++i) xg2[d0 + i][j] = xgT[j][d0 + i];
      }
      __syncthreads();

      // ---- stage 1: S = Wc(64x128) @ xg(128x32), waves 0..7 ----
      if (wid < 8) {
        const int skt = wid >> 1, sjt = wid & 1;
        v8f s = {};
#pragma unroll
        for (int kk = 0; kk < 4; ++kk) {
          v16bf a = load_frag_a(&wc_s[skt * 16][kk * 32], DG);
          v16bf b = load_frag_b(&xgT[sjt * 16][kk * 32], DG);
          s = wmma_bf16(a, b, s);
        }
#pragma unroll
        for (int r = 0; r < 8; ++r) {
          int krow = skt * 16 + r + ((lane < 16) ? 0 : 8);
          int jcol = sjt * 16 + (lane & 15);
          S_s[krow][jcol] = s[r];
        }
      }
      __syncthreads();

      // ---- per-column softmax stats + attention scale ----
      if (tid < VT) {
        const int j = tid, l = l0 + j;
        float m = -1e30f;
        for (int k = 0; k < KK; ++k) m = fmaxf(m, S_s[k][j]);
        float sum = 0.f;
        for (int k = 0; k < KK; ++k) sum += __expf(S_s[k][j] - m);
        float a = (l < LL) ? ap[l] : 0.f;  // a=0 kills padded columns
        cmax[j] = m;
        cscale[j] = a / sum;
      }
      __syncthreads();

      // ---- w = softmax * att -> w_s[k][j]; keep per-thread sum-w ----
      {
        const float m = cmax[myj], sc = cscale[myj];
        float w0 = __expf(S_s[myk][myj] - m) * sc;
        float w1 = __expf(S_s[myk + 32][myj] - m) * sc;
        w_s[myk][myj] = (bf16_t)w0;
        w_s[myk + 32][myj] = (bf16_t)w1;
        ws0 += w0;
        ws1 += w1;
      }
      __syncthreads();

      // ---- stage 2: acc(d,k) += xg(128x32;j) @ w^T(32xj,64k) ----
      {
        v16bf a = load_frag_a(&xg2[dt * 16][0], VT);
        v16bf b = load_frag_b(&w_s[kt * 16][0], VT);
        acc = wmma_bf16(a, b, acc);
      }
      __syncthreads();
    }
  }

  // ---- reduce sum-w partials (ds_add_f32) ----
  atomicAdd(&wsum_s[myk], ws0);
  atomicAdd(&wsum_s[myk + 32], ws1);
  __syncthreads();

  // ---- centroid correction + fp32 store: vlad[n,k,d] ----
#pragma unroll
  for (int r = 0; r < 8; ++r) {
    int d = dt * 16 + r + ((lane < 16) ? 0 : 8);
    int k = kt * 16 + (lane & 15);
    float v = acc[r] - centroids[(size_t)k * DG + d] * wsum_s[k];
    out[((size_t)n * KK + k) * DG + d] = v;
  }
}

// ---------------------------------------------------------------------------
// Host launch
// ---------------------------------------------------------------------------
extern "C" void kernel_launch(void* const* d_in, const int* in_sizes, int n_in,
                              void* d_out, int out_size, void* d_ws,
                              size_t ws_size, hipStream_t stream) {
  const float* x    = (const float*)d_in[0];
  const float* W1   = (const float*)d_in[1];
  const float* W2   = (const float*)d_in[2];
  const float* Wc   = (const float*)d_in[3];
  const float* cent = (const float*)d_in[4];
  float* out        = (float*)d_out;

  // workspace layout (16B-aligned offsets)
  char* ws = (char*)d_ws;
  bf16_t* xhatT = (bf16_t*)(ws);               // 32*1216*512*2  = 39,845,888
  bf16_t* W1b   = (bf16_t*)(ws + 39845888);    // 1024*512*2     =  1,048,576
  bf16_t* Wcb   = (bf16_t*)(ws + 40894464);    // 64*128*2       =     16,384
  bf16_t* ybT   = (bf16_t*)(ws + 40910848);    // 32*1216*1024*2 = 79,691,776
  float*  attb  = (float*) (ws + 120602624);   // 32*8*1200*4    =  1,228,800

  k_cvt_bf16<<<(EE * CC + 255) / 256, 256, 0, stream>>>(W1, W1b, EE * CC);
  k_cvt_bf16<<<(KK * DG + 255) / 256, 256, 0, stream>>>(Wc, Wcb, KK * DG);

  k_normalize<<<dim3((LP + 255) / 256, NB), 256, 0, stream>>>(x, xhatT);

  k_gemm1<<<dim3(LP / 64, EE / 128, NB), 256, 0, stream>>>(W1b, xhatT, ybT);

  k_att<<<dim3((LL + 255) / 256, NB), 256, 0, stream>>>(ybT, W2, attb);

  k_vlad<<<NB, 1024, 0, stream>>>(ybT, Wcb, attb, cent, out);
}
